// RgcnLP_43765716746753
// MI455X (gfx1250) — matrix operand order repaired and checked by reading, not compiled, not run on hardware
//
#include <hip/hip_runtime.h>
#include <hip/hip_bf16.h>

// ---------------- problem constants (match reference) ----------------
#define N_ENT 50000
#define N_REL 200
#define DIM   128
#define N_BAS 8
#define N_EDG 160000
#define N_TRI 50000

typedef _Float16 v16h __attribute__((ext_vector_type(16)));
typedef _Float16 v8h  __attribute__((ext_vector_type(8)));
typedef float    v8f  __attribute__((ext_vector_type(8)));
typedef unsigned int v4u __attribute__((ext_vector_type(4)));
typedef int          v8i __attribute__((ext_vector_type(8)));
typedef int          v4i __attribute__((ext_vector_type(4)));

// ---------------- workspace layout (bytes, all 256-aligned) ----------------
// h buffers carry one extra "dump" row (index N_ENT) that absorbs the zero
// contributions of padded tile rows, so the atomic scatter is branch-free.
constexpr size_t CB_OFF   = 0;                                   // combCount: int[N_ENT*N_REL]
constexpr size_t CB_BYTES = (size_t)N_ENT * N_REL * 4;           // 40,000,000
constexpr size_t RH_OFF   = CB_OFF + CB_BYTES;                   // relHist: int[200]
constexpr size_t RC_OFF   = RH_OFF + 1024;                       // relCursor: int[200]
constexpr size_t HP_BYTES = (size_t)(N_ENT + 1) * DIM * 4;       // 25,600,512 (padded)
constexpr size_t H1_OFF   = RC_OFF + 1024;                       // h1: float[(N_ENT+1)*DIM]
constexpr size_t H2_OFF   = H1_OFF + HP_BYTES;                   // h2: float[(N_ENT+1)*DIM]
constexpr size_t ZERO_BYTES = H2_OFF + HP_BYTES;                 // region zeroed per call
constexpr size_t RS_OFF   = ZERO_BYTES;                          // relStart: int[201]
constexpr size_t SE_OFF   = RS_OFF + 1024;                       // sortedEdge: int[N_EDG]
constexpr size_t WH_BYTES = (size_t)N_REL * DIM * DIM * 2;       // 6,553,600 (f16)
constexpr size_t W1_OFF   = SE_OFF + (size_t)N_EDG * 4;
constexpr size_t W2_OFF   = W1_OFF + WH_BYTES;
constexpr size_t XH_OFF   = W2_OFF + WH_BYTES;                   // xh: f16[N_ENT*DIM] (per-layer features)

// ---------------- kernels ----------------

// Per-edge histogram: edges per (dst,rel) pair and edges per relation.
__global__ void count_edges(const int* __restrict__ edge_index,
                            const int* __restrict__ edge_attr,
                            int* __restrict__ combCount,
                            int* __restrict__ relHist) {
  int e = blockIdx.x * blockDim.x + threadIdx.x;
  if (e >= N_EDG) return;
  int dst = edge_index[N_EDG + e];
  int r   = edge_attr[e];
  atomicAdd(&combCount[(size_t)dst * N_REL + r], 1);
  atomicAdd(&relHist[r], 1);
}

// Tiny exclusive scan over the 200 relation bins.
__global__ void scan_rel(const int* __restrict__ relHist, int* __restrict__ relStart) {
  if (blockIdx.x == 0 && threadIdx.x == 0) {
    int acc = 0;
    for (int i = 0; i < N_REL; ++i) { relStart[i] = acc; acc += relHist[i]; }
    relStart[N_REL] = acc;
  }
}

// Counting-sort scatter: edge ids grouped by relation.
__global__ void scatter_edges(const int* __restrict__ edge_attr,
                              const int* __restrict__ relStart,
                              int* __restrict__ relCursor,
                              int* __restrict__ sortedEdge) {
  int e = blockIdx.x * blockDim.x + threadIdx.x;
  if (e >= N_EDG) return;
  int r = edge_attr[e];
  int pos = relStart[r] + atomicAdd(&relCursor[r], 1);
  sortedEdge[pos] = e;
}

// W[r] = sum_b coef[r,b] * bases[b]  ->  f16
__global__ void make_w(const float* __restrict__ bases,
                       const float* __restrict__ coef,
                       _Float16* __restrict__ Wh) {
  long idx = (long)blockIdx.x * blockDim.x + threadIdx.x;
  if (idx >= (long)N_REL * DIM * DIM) return;
  int r   = (int)(idx >> 14);          // / (128*128)
  int rem = (int)(idx & 16383);        // d*128 + o
  float acc = 0.f;
  #pragma unroll
  for (int b = 0; b < N_BAS; ++b)
    acc += coef[r * N_BAS + b] * bases[(size_t)b * DIM * DIM + rem];
  Wh[idx] = (_Float16)acc;
}

// Feature convert: f32 -> f16 with optional leaky_relu (slope=1 => identity).
__global__ void cvt_f16_k(const float* __restrict__ in, _Float16* __restrict__ out,
                          float slope, long n) {
  long i = (long)blockIdx.x * blockDim.x + threadIdx.x;
  if (i >= n) return;
  float v = in[i];
  v = v > 0.f ? v : slope * v;
  out[i] = (_Float16)v;
}

// grid = (N_REL, REL_SPLIT). Each block: one relation, 1/REL_SPLIT of its edge
// tiles, 3 wave32s. W[r] staged in LDS ([out][in]); the 16-row A gather runs on
// the Tensor Data Mover (gather-mode D#, 16-bit row indices) into a double-
// buffered LDS tile pipelined with TENSORcnt; compute is v_wmma_f32_16x16x32_f16;
// scatter is branch-free atomic f32 adds with 1/count folded into the epilogue.
#define NWAVES 3
#define REL_SPLIT 6
__global__ __launch_bounds__(32 * NWAVES)
void rgcn_layer_wmma(const _Float16* __restrict__ xh,        // [N_ENT,DIM] f16
                     const _Float16* __restrict__ Wh,        // [N_REL,DIM,DIM] (d-major)
                     const int* __restrict__ edge_index,     // [2,N_EDG]
                     const int* __restrict__ sortedEdge,
                     const int* __restrict__ relStart,
                     const int* __restrict__ combCount,
                     float* __restrict__ hOut)               // [(N_ENT+1),DIM], pre-zeroed
{
  __shared__ __align__(16) _Float16 Wlds[DIM * DIM];            // 32KB, [o][d]
  __shared__ __align__(16) _Float16 Abuf[NWAVES][2][16 * DIM];  // 24KB, TDM double buffer
  __shared__ int   metaD[NWAVES][2][16];                        // per-row dst
  __shared__ float metaI[NWAVES][2][16];                        // per-row 1/count

  const int r    = blockIdx.x;
  const int tid  = threadIdx.x;
  const int wave = tid >> 5;
  const int lane = tid & 31;

  // Stage W[r] into LDS, transposing [d][o] -> [o][d] so B-fragments are
  // pairs of contiguous 8-half runs 16 apart (ds_load_b128 each).
  const _Float16* Wg = Wh + (size_t)r * DIM * DIM;
  for (int i = tid; i < DIM * DIM; i += 32 * NWAVES) {
    int d = i >> 7, o = i & 127;
    Wlds[o * DIM + d] = Wg[i];
  }
  __syncthreads();

  const int eBeg   = relStart[r];
  const int cnt    = relStart[r + 1] - eBeg;
  const int ntileE = (cnt + 15) >> 4;
  const int tStep  = NWAVES * REL_SPLIT;
  const int myT0   = wave + NWAVES * blockIdx.y;

  const int mrow   = lane & 15;               // A row this lane owns
  const int kbA    = (lane < 16) ? 0 : 8;     // A K-sub-base per ISA layout
  const int kbB    = (lane < 16) ? 0 : 8;     // B K-sub-base per ISA layout
  const int ncol   = lane & 15;               // B/C column this lane owns

  if (myT0 >= ntileE) return;

  const unsigned ldsA0 = (unsigned)(uintptr_t)&Abuf[wave][0][0];
  const unsigned ldsA1 = (unsigned)(uintptr_t)&Abuf[wave][1][0];

  // Gather a tile's edge metadata (lanes 0-15 in parallel), park dst/inv in
  // LDS, pack the 16 source row ids via readlane, and issue the TDM gather.
  auto stage = [&](int tt, int buf) {
    int s = 0;
    if (lane < 16) {
      int ei = tt * 16 + lane;
      int d = N_ENT; float inv = 0.f;          // pad: gather row 0, scale 0, dump row
      if (ei < cnt) {
        int e = sortedEdge[eBeg + ei];
        s = edge_index[e];
        d = edge_index[N_EDG + e];
        int c = combCount[(size_t)d * N_REL + r];
        inv = 1.0f / (float)(c > 1 ? c : 1);
      }
      metaD[wave][buf][lane] = d;
      metaI[wave][buf][lane] = inv;
    }
    // pack 16 x u16 row indices into 8 dwords (uniform, via readlane)
    unsigned pk[8];
    #pragma unroll
    for (int j = 0; j < 8; ++j) {
      unsigned lo = (unsigned)__builtin_amdgcn_readlane(s, 2 * j) & 0xFFFFu;
      unsigned hi = (unsigned)__builtin_amdgcn_readlane(s, 2 * j + 1);
      pk[j] = lo | (hi << 16);
    }
    // D# group 0: count=1, gather_mode=1, 16-bit indices, lds_addr, global_addr, type=2
    unsigned long long ga = (unsigned long long)(uintptr_t)xh;
    v4u g0;
    g0[0] = 0x80000001u;
    g0[1] = buf ? ldsA1 : ldsA0;
    g0[2] = (unsigned)ga;
    g0[3] = ((unsigned)(ga >> 32) & 0x01FFFFFFu) | 0x80000000u;
    // D# group 1: data_size=2B, tensor 128 x N_ENT (stride 128), tile 128 x 16 idx
    v8i g1;
    g1[0] = 1 << 16;                       // data_size code 1 (2 bytes)
    g1[1] = DIM << 16;                     // tensor_dim0[15:0] = 128
    g1[2] = (int)((N_ENT & 0xFFFF) << 16); // tensor_dim1[15:0]
    g1[3] = DIM << 16;                     // tile_dim0 = 128
    g1[4] = 16;                            // tile_dim1 = #valid indices
    g1[5] = DIM;                           // tensor_dim0_stride = 128
    g1[6] = 0;
    g1[7] = 0;
    v4i g2, g3;
    g2[0] = (int)pk[0]; g2[1] = (int)pk[1]; g2[2] = (int)pk[2]; g2[3] = (int)pk[3];
    g3[0] = (int)pk[4]; g3[1] = (int)pk[5]; g3[2] = (int)pk[6]; g3[3] = (int)pk[7];
    v8i g4 = {0, 0, 0, 0, 0, 0, 0, 0};     // clang-23 extra group, unused
    __builtin_amdgcn_tensor_load_to_lds(g0, g1, g2, g3, g4, 0);
  };

  // prologue: async-gather first tile into buffer 0
  stage(myT0, 0);

  int p = 0;
  for (int t = myT0; t < ntileE; t += tStep) {
    // ---- issue next tile's TDM gather into the other buffer ----
    int tn = t + tStep;
    stage((tn < ntileE) ? tn : t, p ^ 1);      // clamp: harmless dummy reload

    // ---- wait until current buffer's gather has landed in LDS ----
    __builtin_amdgcn_s_wait_tensorcnt(1);

    // ---- A fragments (N-tile invariant) from the current buffer ----
    const _Float16* Ab = &Abuf[wave][p][0];
    v16h afrag[4];
    #pragma unroll
    for (int kc = 0; kc < 4; ++kc) {
      const _Float16* arow = Ab + mrow * DIM + kc * 32 + kbA;
      v8h alo = *(const v8h*)(arow);        // K = kbA+0..7
      v8h ahi = *(const v8h*)(arow + 16);   // K = kbA+16..23
      afrag[kc] = __builtin_shufflevector(alo, ahi,
                  0,1,2,3,4,5,6,7,8,9,10,11,12,13,14,15);
    }

    // ---- per-row scatter addresses + 1/count scales (from LDS meta) ----
    const int mbaseC = (lane < 16) ? 0 : 8;   // C row base per ISA layout
    float* addr[8]; float invSel[8];
    #pragma unroll
    for (int v = 0; v < 8; ++v) {
      int   d   = metaD[wave][p][mbaseC + v];
      invSel[v] = metaI[wave][p][mbaseC + v];
      addr[v] = hOut + (size_t)d * DIM + ncol;
    }

    // ---- 8 N-tiles x 4 K-chunks of v_wmma_f32_16x16x32_f16 ----
    #pragma unroll
    for (int nt = 0; nt < 8; ++nt) {
      v8f c = {};
      #pragma unroll
      for (int kc = 0; kc < 4; ++kc) {
        // B row = output column (nt*16 + ncol) in [o][d]-transposed LDS
        const _Float16* brow = &Wlds[(nt * 16 + ncol) * DIM + kc * 32 + kbB];
        v8h blo = *(const v8h*)(brow);
        v8h bhi = *(const v8h*)(brow + 16);
        v16h b = __builtin_shufflevector(blo, bhi,
                 0,1,2,3,4,5,6,7,8,9,10,11,12,13,14,15);

        c = __builtin_amdgcn_wmma_f32_16x16x32_f16(
                false, afrag[kc], false, b, (short)0, c, false, false);
      }
      // branch-free scatter; normalization folded here; pads add 0 to dump row
      #pragma unroll
      for (int v = 0; v < 8; ++v)
        unsafeAtomicAdd(addr[v] + nt * 16, c[v] * invSel[v]);
    }

    p ^= 1;
  }
}

// One wave32 per triple: sigmoid(sum_d h[head]*rel*h[tail]).
__global__ __launch_bounds__(256)
void score_k(const float* __restrict__ h2, const float* __restrict__ rel_embeds,
             const int* __restrict__ triples, float* __restrict__ out) {
  int gtid = blockIdx.x * 256 + threadIdx.x;
  int wi   = gtid >> 5;
  int lane = gtid & 31;
  if (wi >= N_TRI) return;
  int head = triples[wi * 3 + 0];
  int rel  = triples[wi * 3 + 1];
  int tail = triples[wi * 3 + 2];
  const float* hh = h2 + (size_t)head * DIM;
  const float* rr = rel_embeds + (size_t)rel * DIM;
  const float* tt = h2 + (size_t)tail * DIM;
  float s = 0.f;
  #pragma unroll
  for (int j = 0; j < DIM / 32; ++j) {
    int d = lane + 32 * j;
    s += hh[d] * rr[d] * tt[d];
  }
  #pragma unroll
  for (int o = 16; o > 0; o >>= 1) s += __shfl_down(s, o, 32);
  if (lane == 0) out[wi] = 1.0f / (1.0f + expf(-s));
}

// ---------------- host launcher ----------------
extern "C" void kernel_launch(void* const* d_in, const int* in_sizes, int n_in,
                              void* d_out, int out_size, void* d_ws, size_t ws_size,
                              hipStream_t stream) {
  const float* ent_embeds = (const float*)d_in[0];
  const float* bases1     = (const float*)d_in[1];
  const float* coef1      = (const float*)d_in[2];
  const float* bases2     = (const float*)d_in[3];
  const float* coef2      = (const float*)d_in[4];
  const float* rel_embeds = (const float*)d_in[5];
  const int*   edge_index = (const int*)d_in[6];
  const int*   edge_attr  = (const int*)d_in[7];
  const int*   triples    = (const int*)d_in[8];
  float* out = (float*)d_out;

  char* ws = (char*)d_ws;
  int*      combCount  = (int*)(ws + CB_OFF);
  int*      relHist    = (int*)(ws + RH_OFF);
  int*      relCursor  = (int*)(ws + RC_OFF);
  float*    h1         = (float*)(ws + H1_OFF);
  float*    h2         = (float*)(ws + H2_OFF);
  int*      relStart   = (int*)(ws + RS_OFF);
  int*      sortedEdge = (int*)(ws + SE_OFF);
  _Float16* W1h        = (_Float16*)(ws + W1_OFF);
  _Float16* W2h        = (_Float16*)(ws + W2_OFF);
  _Float16* xh         = (_Float16*)(ws + XH_OFF);

  // zero: combCount, relHist, relCursor, h1, h2 (contiguous region)
  (void)hipMemsetAsync(ws, 0, ZERO_BYTES, stream);

  // counting sort by relation + per-(dst,rel) counts
  count_edges<<<(N_EDG + 255) / 256, 256, 0, stream>>>(edge_index, edge_attr, combCount, relHist);
  scan_rel<<<1, 32, 0, stream>>>(relHist, relStart);
  scatter_edges<<<(N_EDG + 255) / 256, 256, 0, stream>>>(edge_attr, relStart, relCursor, sortedEdge);

  // relation weight tensors in f16
  long wElems = (long)N_REL * DIM * DIM;
  int  wBlk   = (int)((wElems + 255) / 256);
  make_w<<<wBlk, 256, 0, stream>>>(bases1, coef1, W1h);
  make_w<<<wBlk, 256, 0, stream>>>(bases2, coef2, W2h);

  long hElems = (long)N_ENT * DIM;
  int  hBlk   = (int)((hElems + 255) / 256);
  dim3 lgrid(N_REL, REL_SPLIT);

  // layer 1: features -> f16, WMMA+TDM layer
  cvt_f16_k<<<hBlk, 256, 0, stream>>>(ent_embeds, xh, 1.0f, hElems);
  rgcn_layer_wmma<<<lgrid, 32 * NWAVES, 0, stream>>>(
      xh, W1h, edge_index, sortedEdge, relStart, combCount, h1);

  // layer 2: leaky_relu folded into the f16 convert
  cvt_f16_k<<<hBlk, 256, 0, stream>>>(h1, xh, 0.01f, hElems);
  rgcn_layer_wmma<<<lgrid, 32 * NWAVES, 0, stream>>>(
      xh, W2h, edge_index, sortedEdge, relStart, combCount, h2);

  // DistMult scoring, one wave per triple
  int scoreBlocks = (N_TRI * 32 + 255) / 256;
  score_k<<<scoreBlocks, 256, 0, stream>>>(h2, rel_embeds, triples, out);
}
